// TranscendentalMetaRL_5540507812452
// MI455X (gfx1250) — matrix operand order repaired and checked by reading, not compiled, not run on hardware
//
#include <hip/hip_runtime.h>

// ---------------------------------------------------------------------------
// TranscendentalMetaRL for MI455X (gfx1250, wave32, WMMA bf16 -> f32).
//
// Math notes (exact transformations):
//  * enh = (1/L) * sum_l lw[b,l] * (x @ cons_W[l]^T + cons_b[l])
//        = x @ W_eff[b]^T + b_eff[b]      (linearity fold, 5x fewer FLOPs)
//  * scores += 0.1*phase[b,h] is constant along the softmax axis -> cancels.
//    freq_W / freq_b / phi_phase / pooled are mathematically dead.
//  * aw2 = softmax(f * softmax(s)), f = factor[b,h] scalar:
//    max_k softmax(s)_k = 1/Z, so two-pass flash: pass1 -> (m, Z);
//    pass2 -> t = exp(f*(p - 1/Z)), attended = (sum t*v) / (sum t).
//
// Data movement: GLOBAL_LOAD_ASYNC_TO_LDS_B128 (ASYNCcnt) stages all tiles
// directly into LDS; the WMMA GEMM double-buffers LDS so the DMA for slab n+1
// overlaps the v_wmma work on slab n. Precision: bf16 in, f32 accumulate.
// ---------------------------------------------------------------------------

#define Bz 2
#define Sq 2048
#define Ee 512
#define Hh 8
#define HDd 64
#define Ll 5

typedef unsigned short bf16_t;
typedef __attribute__((ext_vector_type(16))) __bf16 v16bf;
typedef __attribute__((ext_vector_type(8)))  float  v8f;

static __device__ __forceinline__ bf16_t f2bf(float f) {
  union { float f; unsigned u; } c; c.f = f;
  unsigned r = c.u + 0x7FFFu + ((c.u >> 16) & 1u);   // round-to-nearest-even
  return (bf16_t)(r >> 16);
}

// ---- CDNA5 async Global->LDS copy (16B per lane), tracked by ASYNCcnt ------
// VDST VGPR = LDS byte address (low 32 bits of generic shared pointer),
// VADDR = 64-bit global address.  See CDNA5 ISA.
static __device__ __forceinline__ void async_ld16(void* lds, const void* gaddr) {
  unsigned l = (unsigned)(size_t)lds;
  asm volatile("global_load_async_to_lds_b128 %0, %1, off"
               :: "v"(l), "v"(gaddr) : "memory");
}
#define ASYNC_WAIT(n) asm volatile("s_wait_asynccnt " #n ::: "memory")

union FragU { v16bf v; bf16_t u[16]; };

// A fragment: 16x32 (MxK) bf16, source row-major with leading dim ld.
// lane holds row M=lane%16; lanes 0-15 K=0..15; lanes 16-31
// K=16..31 (matches B fragments). Per-lane runs are contiguous.
static __device__ __forceinline__ v16bf load_a_frag(const bf16_t* t, int ld, int lane) {
  FragU f;
  const int row = lane & 15, kg = (lane >> 4) * 16;
#pragma unroll
  for (int p = 0; p < 8; ++p) {
    int kb = kg + 2 * p;
    f.u[2 * p]     = t[row * ld + kb];
    f.u[2 * p + 1] = t[row * ld + kb + 1];
  }
  return f.v;
}

// B fragment: 32x16 (KxN), SOURCE N-major (rows=N, cols=K) i.e. A @ W^T with
// W stored (N,K) row-major. lane holds col N=lane%16; lanes 0-15 K=0..15.
static __device__ __forceinline__ v16bf load_b_frag_nk(const bf16_t* t, int ld, int lane) {
  FragU f;
  const int col = lane & 15, k0 = (lane >> 4) * 16;
#pragma unroll
  for (int p = 0; p < 8; ++p) {
    f.u[2 * p]     = t[col * ld + k0 + 2 * p];
    f.u[2 * p + 1] = t[col * ld + k0 + 2 * p + 1];
  }
  return f.v;
}

// B fragment: 32x16 (KxN) with SOURCE K-major (rows=K, cols=N), e.g. V.
static __device__ __forceinline__ v16bf load_b_frag_kn(const bf16_t* t, int ld, int lane) {
  FragU f;
  const int col = lane & 15, k0 = (lane >> 4) * 16;
#pragma unroll
  for (int p = 0; p < 8; ++p) {
    f.u[2 * p]     = t[(k0 + 2 * p) * ld + col];
    f.u[2 * p + 1] = t[(k0 + 2 * p + 1) * ld + col];
  }
  return f.v;
}

#define WMMA_BF16(a, b, c) \
  __builtin_amdgcn_wmma_f32_16x16x32_bf16(false, (a), false, (b), (short)0, (c), false, false)

// ---------------------------------------------------------------------------
// f32 -> bf16 conversion
__global__ void cvt_bf16(const float* __restrict__ src, bf16_t* __restrict__ dst, int n) {
  int i = blockIdx.x * blockDim.x + threadIdx.x;
  if (i < n) dst[i] = f2bf(src[i]);
}

// ---------------------------------------------------------------------------
// prep: lw, gw=softmax(gate,axis=1), factor, b_eff (one small block)
__global__ __launch_bounds__(256) void prep(
    const float* __restrict__ cons_levels, const float* __restrict__ gate,
    const float* __restrict__ cons_b,
    float* __restrict__ lw, float* __restrict__ factor, float* __restrict__ b_eff) {
  const int tid = threadIdx.x;
  __shared__ float lws[Bz * Ll], gws[Ll * Hh];
  if (tid < Bz * Ll) {
    int b = tid / Ll, l = tid % Ll;
    float v = cons_levels[b * Hh + (l % Hh)];
    lws[tid] = v; lw[tid] = v;
  }
  if (tid < Ll) {
    float mx = -1e30f;
    for (int hh = 0; hh < Hh; ++hh) mx = fmaxf(mx, gate[tid * Hh + hh]);
    float s = 0.f, e[Hh];
    for (int hh = 0; hh < Hh; ++hh) { e[hh] = __expf(gate[tid * Hh + hh] - mx); s += e[hh]; }
    for (int hh = 0; hh < Hh; ++hh) gws[tid * Hh + hh] = e[hh] / s;
  }
  __syncthreads();
  if (tid < Bz * Hh) {
    int b = tid / Hh, hh = tid % Hh;
    float fp = 1.f;
    for (int l = 0; l < Ll; ++l) fp *= (1.f + 0.1f * lws[b * Ll + l] * gws[l * Hh + hh]);
    factor[tid] = fp;
  }
  for (int i = tid; i < Bz * Ee; i += 256) {
    int b = i / Ee, e = i % Ee;
    float s = 0.f;
    for (int l = 0; l < Ll; ++l) s += lws[b * Ll + l] * cons_b[l * Ee + e];
    b_eff[i] = s * (1.f / Ll);
  }
}

// ---------------------------------------------------------------------------
// W_eff[b] = (1/L) sum_l lw[b,l] * cons_W[l]   (f32 -> bf16)
__global__ __launch_bounds__(256) void weff(
    const float* __restrict__ cons_W, const float* __restrict__ lw,
    bf16_t* __restrict__ Weff) {
  size_t idx = ((size_t)blockIdx.x * 256 + threadIdx.x) * 4;
  const size_t per = (size_t)Ee * Ee;
  int b = (int)(idx / per);
  size_t r = idx % per;
  float ax = 0, ay = 0, az = 0, aw = 0;
  for (int l = 0; l < Ll; ++l) {
    float w = lw[b * Ll + l] * (1.f / Ll);
    float4 cw = reinterpret_cast<const float4*>(cons_W + (size_t)l * per)[r >> 2];
    ax += w * cw.x; ay += w * cw.y; az += w * cw.z; aw += w * cw.w;
  }
  bf16_t* o = Weff + (size_t)b * per + r;
  o[0] = f2bf(ax); o[1] = f2bf(ay); o[2] = f2bf(az); o[3] = f2bf(aw);
}

// ---------------------------------------------------------------------------
// Generic WMMA GEMM: C[m,n] = sum_k A[m,k] * W[n,k] + bias[n]  (+ epilogue)
// Block 256 threads (8 waves), tile 64(M) x 128(N), BK=32, wave tile 32x32.
// LDS double-buffered; tiles staged with GLOBAL_LOAD_ASYNC_TO_LDS_B128 so the
// DMA for slab n+1 overlaps the WMMAs on slab n (3 async ops / thread / tile).
// mode 0: out bf16 row-major (b,M,N)      [enh]
// mode 1: out bf16 head layout (b,h,s,d)  [q/k/v]
// mode 2: out f32  row-major + residual   [y = x + attended@Wo^T + bo]
__global__ __launch_bounds__(256) void gemm_nt(
    const bf16_t* __restrict__ Aall, const bf16_t* __restrict__ Wall,
    const float* __restrict__ bias, const float* __restrict__ resid,
    void* __restrict__ outp, int mode, int perBatchW, int M, int N, int K) {
  const int b = blockIdx.z;
  const bf16_t* A = Aall + (size_t)b * M * K;
  const bf16_t* W = Wall + (perBatchW ? (size_t)b * N * K : 0);
  const float* bi = bias + (perBatchW ? (size_t)b * N : 0);
  const int n0 = blockIdx.x * 128, m0 = blockIdx.y * 64;
  __shared__ __align__(16) bf16_t As[2][64 * 32];
  __shared__ __align__(16) bf16_t Ws[2][128 * 32];
  const int tid = threadIdx.x, lane = tid & 31, w = tid >> 5;
  const int wm = w & 1, wn = w >> 1;
  v8f acc[2][2] = {};

  // per-thread async staging of one 64x32 A tile + 128x32 W tile
  const int ra = tid >> 2, ca = (tid & 3) * 8;              // A: 1 op
  const int rw0 = tid >> 2, rw1 = (tid + 256) >> 2;         // W: 2 ops
  auto issueTile = [&](int k0, int buf) {
    async_ld16(&As[buf][ra * 32 + ca], A + (size_t)(m0 + ra) * K + k0 + ca);
    async_ld16(&Ws[buf][rw0 * 32 + ca], W + (size_t)(n0 + rw0) * K + k0 + ca);
    async_ld16(&Ws[buf][rw1 * 32 + ca], W + (size_t)(n0 + rw1) * K + k0 + ca);
  };

  issueTile(0, 0);
  const int T = K / 32;
  for (int it = 0; it < T; ++it) {
    const int cur = it & 1;
    if (it + 1 < T) {
      issueTile((it + 1) * 32, cur ^ 1);
      ASYNC_WAIT(0x3);   // the 3 newest (next tile) may remain in flight
    } else {
      ASYNC_WAIT(0x0);
    }
    __syncthreads();
    v16bf af[2], bf[2];
#pragma unroll
    for (int i = 0; i < 2; ++i) af[i] = load_a_frag(&As[cur][(wm * 32 + i * 16) * 32], 32, lane);
#pragma unroll
    for (int j = 0; j < 2; ++j) bf[j] = load_b_frag_nk(&Ws[cur][(wn * 32 + j * 16) * 32], 32, lane);
#pragma unroll
    for (int i = 0; i < 2; ++i)
#pragma unroll
      for (int j = 0; j < 2; ++j) acc[i][j] = WMMA_BF16(af[i], bf[j], acc[i][j]);
    __syncthreads();   // all waves done with buf[cur] before it is re-filled
  }

  // epilogue: C layout lane l, VGPR p -> M = 8*(l>>4)+p, N = l&15
#pragma unroll
  for (int i = 0; i < 2; ++i)
#pragma unroll
    for (int j = 0; j < 2; ++j) {
      int ncol = n0 + wn * 32 + j * 16 + (lane & 15);
      int mrow = m0 + wm * 32 + i * 16 + ((lane >> 4) << 3);
      float bsv = bi[ncol];
#pragma unroll
      for (int p = 0; p < 8; ++p) {
        int r = mrow + p;
        float vout = acc[i][j][p] + bsv;
        if (mode == 0) {
          ((bf16_t*)outp)[((size_t)b * M + r) * N + ncol] = f2bf(vout);
        } else if (mode == 1) {
          int h = ncol >> 6, d = ncol & 63;
          ((bf16_t*)outp)[(((size_t)b * Hh + h) * Sq + r) * HDd + d] = f2bf(vout);
        } else {
          size_t o = ((size_t)b * M + r) * N + ncol;
          ((float*)outp)[o] = vout + resid[o];
        }
      }
    }
}

// ---------------------------------------------------------------------------
// Two-pass flash attention with double softmax.
// Block = 128 threads (4 waves), 64 queries per block, K streamed in 64-chunks
// via async Global->LDS DMA (4 x b128 per thread per chunk).
__global__ __launch_bounds__(128) void attn(
    const bf16_t* __restrict__ Q, const bf16_t* __restrict__ Kt,
    const bf16_t* __restrict__ V, const float* __restrict__ factor,
    bf16_t* __restrict__ attB) {
  const int bh = blockIdx.y, b = bh >> 3, h = bh & 7;
  const int q0 = blockIdx.x * 64;
  const int tid = threadIdx.x, lane = tid & 31, w = tid >> 5;
  __shared__ __align__(16) bf16_t Qs[64 * 64];
  __shared__ __align__(16) bf16_t KVs[64 * 64];
  __shared__ __align__(16) float  Sb[64 * 64];
  __shared__ __align__(16) bf16_t Tb[64 * 64];
  __shared__ float rowm[64], rowz[64], rowz2[64];

  const bf16_t* Qg = Q + ((size_t)bh * Sq + q0) * HDd;
  const bf16_t* Kg = Kt + (size_t)bh * Sq * HDd;
  const bf16_t* Vg = V + (size_t)bh * Sq * HDd;
  const float scale = 0.125f;  // 1/sqrt(64)

  // Q block (64x64 bf16, contiguous): async DMA, completion folded into the
  // first chunk's ASYNC_WAIT below (same-wave async loads complete in order).
#pragma unroll
  for (int i = tid; i < 512; i += 128) async_ld16(&Qs[i * 8], Qg + i * 8);

  // ---------------- pass 1: per-row (m, Z) over full S ----------------
  float m_r = -1e30f, z_r = 0.f;
  for (int kc = 0; kc < Sq; kc += 64) {
    __syncthreads();  // prior chunk's readers done before overwriting KVs
#pragma unroll
    for (int i = tid; i < 512; i += 128)
      async_ld16(&KVs[i * 8], Kg + (size_t)kc * HDd + i * 8);
    ASYNC_WAIT(0x0);
    __syncthreads();
#pragma unroll
    for (int mt = 0; mt < 4; ++mt) {
      v8f accs = {};
#pragma unroll
      for (int ks = 0; ks < 64; ks += 32) {
        v16bf a  = load_a_frag(Qs + (mt * 16) * 64 + ks, 64, lane);
        v16bf bb = load_b_frag_nk(KVs + (w * 16) * 64 + ks, 64, lane);
        accs = WMMA_BF16(a, bb, accs);
      }
      int col = w * 16 + (lane & 15), rb = mt * 16 + ((lane >> 4) << 3);
#pragma unroll
      for (int p = 0; p < 8; ++p) Sb[(rb + p) * 64 + col] = accs[p] * scale;
    }
    __syncthreads();
    if (tid < 64) {  // thread r owns query row r (stable across chunks)
      float cm = -1e30f;
      for (int c = 0; c < 64; ++c) cm = fmaxf(cm, Sb[tid * 64 + c]);
      float cs = 0.f;
      for (int c = 0; c < 64; ++c) cs += __expf(Sb[tid * 64 + c] - cm);
      float nm = fmaxf(m_r, cm);
      z_r = z_r * __expf(m_r - nm) + cs * __expf(cm - nm);
      m_r = nm;
    }
  }
  __syncthreads();
  if (tid < 64) { rowm[tid] = m_r; rowz[tid] = z_r; rowz2[tid] = 0.f; }
  __syncthreads();
  const float f = factor[bh];

  // ---------------- pass 2: t = exp(f*(p - 1/Z)); acc += t*V ----------------
  v8f accv[4] = {};
  float z2_r = 0.f;
  for (int kc = 0; kc < Sq; kc += 64) {
    __syncthreads();
#pragma unroll
    for (int i = tid; i < 512; i += 128)
      async_ld16(&KVs[i * 8], Kg + (size_t)kc * HDd + i * 8);
    ASYNC_WAIT(0x0);
    __syncthreads();
#pragma unroll
    for (int mt = 0; mt < 4; ++mt) {
      v8f accs = {};
#pragma unroll
      for (int ks = 0; ks < 64; ks += 32) {
        v16bf a  = load_a_frag(Qs + (mt * 16) * 64 + ks, 64, lane);
        v16bf bb = load_b_frag_nk(KVs + (w * 16) * 64 + ks, 64, lane);
        accs = WMMA_BF16(a, bb, accs);
      }
      int col = w * 16 + (lane & 15), rb = mt * 16 + ((lane >> 4) << 3);
#pragma unroll
      for (int p = 0; p < 8; ++p) Sb[(rb + p) * 64 + col] = accs[p] * scale;
    }
    __syncthreads();
    // Kick off the V-chunk DMA, then overlap it with the per-row T build.
#pragma unroll
    for (int i = tid; i < 512; i += 128)
      async_ld16(&KVs[i * 8], Vg + (size_t)kc * HDd + i * 8);
    if (tid < 64) {
      float mm = rowm[tid], zz = rowz[tid], c0 = f / zz;
      for (int c = 0; c < 64; ++c) {
        float p = __expf(Sb[tid * 64 + c] - mm) / zz;  // exact softmax(s)
        float t = __expf(f * p - c0);                  // shifted by max(f*p)=f/Z
        z2_r += t;
        Tb[tid * 64 + c] = f2bf(t);
      }
    }
    ASYNC_WAIT(0x0);
    __syncthreads();
    // P·V: wave w owns HD columns [16w,16w+16), all 4 M tiles, K-chunk=64
#pragma unroll
    for (int mt = 0; mt < 4; ++mt)
#pragma unroll
      for (int ks = 0; ks < 64; ks += 32) {
        v16bf a  = load_a_frag(Tb + (mt * 16) * 64 + ks, 64, lane);
        v16bf bb = load_b_frag_kn(KVs + ks * 64 + w * 16, 64, lane);
        accv[mt] = WMMA_BF16(a, bb, accv[mt]);
      }
  }
  __syncthreads();
  if (tid < 64) rowz2[tid] = z2_r;
  __syncthreads();
#pragma unroll
  for (int mt = 0; mt < 4; ++mt) {
    int rb = mt * 16 + ((lane >> 4) << 3), d = w * 16 + (lane & 15);
#pragma unroll
    for (int p = 0; p < 8; ++p) {
      int r = rb + p;
      float vout = accv[mt][p] / rowz2[r];
      attB[((size_t)b * Sq + q0 + r) * Ee + h * HDd + d] = f2bf(vout);
    }
  }
}

// ---------------------------------------------------------------------------
// Row LayerNorm: out = (y-mu)/sqrt(var+eps)*g + b, row length E=512.
__global__ __launch_bounds__(128) void lnorm(
    const float* __restrict__ y, const float* __restrict__ g,
    const float* __restrict__ be, float* __restrict__ out) {
  const int row = blockIdx.x, tid = threadIdx.x;
  const float* yr = y + (size_t)row * Ee;
  float4 v = reinterpret_cast<const float4*>(yr)[tid];
  float s = v.x + v.y + v.z + v.w;
  float s2 = v.x * v.x + v.y * v.y + v.z * v.z + v.w * v.w;
  __shared__ float red0[128], red1[128];
  red0[tid] = s; red1[tid] = s2;
  __syncthreads();
  for (int st = 64; st > 0; st >>= 1) {
    if (tid < st) { red0[tid] += red0[tid + st]; red1[tid] += red1[tid + st]; }
    __syncthreads();
  }
  float mu = red0[0] * (1.f / Ee);
  float var = red1[0] * (1.f / Ee) - mu * mu;
  float inv = rsqrtf(var + 1e-5f);
  float4 gg = reinterpret_cast<const float4*>(g)[tid];
  float4 bb = reinterpret_cast<const float4*>(be)[tid];
  float4 o;
  o.x = (v.x - mu) * inv * gg.x + bb.x;
  o.y = (v.y - mu) * inv * gg.y + bb.y;
  o.z = (v.z - mu) * inv * gg.z + bb.z;
  o.w = (v.w - mu) * inv * gg.w + bb.w;
  reinterpret_cast<float4*>(out + (size_t)row * Ee)[tid] = o;
}

// ---------------------------------------------------------------------------
extern "C" void kernel_launch(void* const* d_in, const int* in_sizes, int n_in,
                              void* d_out, int out_size, void* d_ws, size_t ws_size,
                              hipStream_t stream) {
  const float* x           = (const float*)d_in[0];
  const float* cons_levels = (const float*)d_in[1];
  const float* cons_W      = (const float*)d_in[2];
  const float* cons_b      = (const float*)d_in[3];
  // d_in[4..5] freq_W/freq_b, d_in[15] phi_phase: mathematically dead (see top)
  const float* Wq = (const float*)d_in[6];  const float* bq = (const float*)d_in[7];
  const float* Wk = (const float*)d_in[8];  const float* bk = (const float*)d_in[9];
  const float* Wv = (const float*)d_in[10]; const float* bv = (const float*)d_in[11];
  const float* Wo = (const float*)d_in[12]; const float* bo = (const float*)d_in[13];
  const float* gate = (const float*)d_in[14];
  const float* ln_g = (const float*)d_in[16];
  const float* ln_b = (const float*)d_in[17];
  float* out = (float*)d_out;

  char* ws = (char*)d_ws;
  size_t off = 0;
  auto alloc = [&](size_t bytes) -> void* {
    void* p = ws + off;
    off += (bytes + 255) & ~(size_t)255;
    return p;
  };
  const size_t NE = (size_t)Bz * Sq * Ee;          // 2M elems
  float*  lw     = (float*)alloc(Bz * Ll * 4);
  float*  factor = (float*)alloc(Bz * Hh * 4);
  float*  b_eff  = (float*)alloc((size_t)Bz * Ee * 4);
  bf16_t* Weff   = (bf16_t*)alloc((size_t)Bz * Ee * Ee * 2);
  bf16_t* xb     = (bf16_t*)alloc(NE * 2);
  bf16_t* wqb    = (bf16_t*)alloc((size_t)Ee * Ee * 2);
  bf16_t* wkb    = (bf16_t*)alloc((size_t)Ee * Ee * 2);
  bf16_t* wvb    = (bf16_t*)alloc((size_t)Ee * Ee * 2);
  bf16_t* wob    = (bf16_t*)alloc((size_t)Ee * Ee * 2);
  bf16_t* enh    = (bf16_t*)alloc(NE * 2);
  bf16_t* qb     = (bf16_t*)alloc(NE * 2);
  bf16_t* kb     = (bf16_t*)alloc(NE * 2);
  bf16_t* vb     = (bf16_t*)alloc(NE * 2);
  bf16_t* attB   = (bf16_t*)alloc(NE * 2);
  float*  yb     = (float*)alloc(NE * 4);
  (void)in_sizes; (void)n_in; (void)out_size; (void)ws_size;

  const int nEE = Ee * Ee;
  cvt_bf16<<<(int)((NE + 255) / 256), 256, 0, stream>>>(x, xb, (int)NE);
  cvt_bf16<<<(nEE + 255) / 256, 256, 0, stream>>>(Wq, wqb, nEE);
  cvt_bf16<<<(nEE + 255) / 256, 256, 0, stream>>>(Wk, wkb, nEE);
  cvt_bf16<<<(nEE + 255) / 256, 256, 0, stream>>>(Wv, wvb, nEE);
  cvt_bf16<<<(nEE + 255) / 256, 256, 0, stream>>>(Wo, wob, nEE);
  prep<<<1, 256, 0, stream>>>(cons_levels, gate, cons_b, lw, factor, b_eff);
  weff<<<(Bz * nEE) / 1024, 256, 0, stream>>>(cons_W, lw, Weff);

  dim3 gg(Ee / 128, Sq / 64, Bz);
  // enh = x @ W_eff[b]^T + b_eff[b]   (per-batch weight+bias)
  gemm_nt<<<gg, 256, 0, stream>>>(xb, Weff, b_eff, nullptr, enh, 0, 1, Sq, Ee, Ee);
  // q/k/v projections into (b,h,s,d) bf16
  gemm_nt<<<gg, 256, 0, stream>>>(enh, wqb, bq, nullptr, qb, 1, 0, Sq, Ee, Ee);
  gemm_nt<<<gg, 256, 0, stream>>>(enh, wkb, bk, nullptr, kb, 1, 0, Sq, Ee, Ee);
  gemm_nt<<<gg, 256, 0, stream>>>(enh, wvb, bv, nullptr, vb, 1, 0, Sq, Ee, Ee);

  dim3 ga(Sq / 64, Bz * Hh);
  attn<<<ga, 128, 0, stream>>>(qb, kb, vb, factor, attB);

  // y = x + attended @ Wo^T + bo
  gemm_nt<<<gg, 256, 0, stream>>>(attB, wob, bo, x, yb, 2, 0, Sq, Ee, Ee);
  lnorm<<<Bz * Sq, 128, 0, stream>>>(yb, ln_g, ln_b, out);
}
